// BaseTraceModel_26508538151736
// MI455X (gfx1250) — compile-verified, matrix operand order checked by reading, not run on hardware
//
#include <hip/hip_runtime.h>
#include <hip/hip_bf16.h>

typedef __attribute__((ext_vector_type(16))) _Float16 v16h;
typedef __attribute__((ext_vector_type(8)))  _Float16 v8h;
typedef __attribute__((ext_vector_type(8)))  float    v8f;

#define B_TOT    8192
#define T_OBS_N  128
#define T_FUT_N  64
#define DDIM     5
#define HDIM     128
#define G3       (3 * HDIM)     // 384
#define BTILE    32
#define NTHREADS 256
#define H16_PITCH 136           // halfs; 272B row stride -> conflict-free b128 column reads
#define H32_PITCH 132           // floats; 4-bank skew per row
#define X_PITCH   8

__device__ __forceinline__ float sigmoid_f(float x) { return 1.0f / (1.0f + __expf(-x)); }
__device__ __forceinline__ float tanh_f(float x)    { return 1.0f - 2.0f / (__expf(2.0f * x) + 1.0f); }

__global__ __launch_bounds__(NTHREADS) void gru_seq2seq_kernel(
    const float* __restrict__ obs,  const float* __restrict__ target,
    const float* __restrict__ eWih, const float* __restrict__ eWhh,
    const float* __restrict__ ebih, const float* __restrict__ ebhh,
    const float* __restrict__ cWih, const float* __restrict__ cWhh,
    const float* __restrict__ cbih, const float* __restrict__ cbhh,
    const float* __restrict__ headW, const float* __restrict__ headb,
    float* __restrict__ out)
{
    __shared__ __align__(16) _Float16 sH16[BTILE][H16_PITCH]; // h in f16 (WMMA A feed)
    __shared__ float    sH32[BTILE][H32_PITCH];               // h in f32 (master)
    __shared__ float    sX[BTILE][X_PITCH];                   // current-step inputs
    __shared__ float    sWih[G3 * DDIM];
    __shared__ float    sBih[G3];
    __shared__ float    sBhh[G3];
    __shared__ float    sHW[DDIM * HDIM];
    __shared__ float    sHB[DDIM];

    const int tid   = threadIdx.x;
    const int lane  = tid & 31;
    const int wave  = tid >> 5;          // 0..7 -> column group (16 hidden units)
    const int lidx  = lane & 15;
    const int mh    = (lane < 16) ? 0 : 8;   // C/D layout: row half
    const int koff  = (lane < 16) ? 0 : 8;   // A/B layout: K sub-offset
    const int bbase = blockIdx.x * BTILE;
    const int jcol  = wave * 16 + lidx;      // hidden unit owned by this lane

    // ---------------- prologue: zero h, stage weights, x(0) ----------------
    for (int i = tid; i < BTILE * HDIM; i += NTHREADS) {
        sH16[i >> 7][i & 127] = (_Float16)0.0f;
        sH32[i >> 7][i & 127] = 0.0f;
    }
    for (int i = tid; i < DDIM * HDIM; i += NTHREADS) sHW[i] = headW[i];
    if (tid < DDIM) sHB[tid] = headb[tid];
    for (int i = tid; i < G3 * DDIM; i += NTHREADS) sWih[i] = eWih[i];
    for (int i = tid; i < G3; i += NTHREADS) { sBih[i] = ebih[i]; sBhh[i] = ebhh[i]; }
    for (int i = tid; i < BTILE * DDIM; i += NTHREADS) {
        int r = i / DDIM, d = i - r * DDIM;
        sX[r][d] = obs[((size_t)(bbase + r) * T_OBS_N + 0) * DDIM + d];
    }
    __syncthreads();

    // Per-lane resident state: input weights/biases for column jcol, and the
    // three gates' B-matrix WMMA fragments (kept in VGPRs for a whole phase).
    float wR[DDIM], wZ[DDIM], wN[DDIM];
    float biR, biZ, biN, bhR, bhZ, bhN;
    v16h Bf[3][4];

    auto lane_setup = [&](const float* __restrict__ Whh) {
#pragma unroll
        for (int d = 0; d < DDIM; ++d) {
            wR[d] = sWih[jcol * DDIM + d];
            wZ[d] = sWih[(HDIM + jcol) * DDIM + d];
            wN[d] = sWih[(2 * HDIM + jcol) * DDIM + d];
        }
        biR = sBih[jcol]; biZ = sBih[HDIM + jcol]; biN = sBih[2 * HDIM + jcol];
        bhR = sBhh[jcol]; bhZ = sBhh[HDIM + jcol]; bhN = sBhh[2 * HDIM + jcol];
#pragma unroll
        for (int g = 0; g < 3; ++g) {
            const float* wrow = Whh + (size_t)(g * HDIM + jcol) * HDIM; // B[k][n] = Whh[n][k]
#pragma unroll
            for (int c = 0; c < 4; ++c) {
                int kb = c * 32 + koff;
                v16h b;
#pragma unroll
                for (int i = 0; i < 8; ++i) {
                    b[i]     = (_Float16)wrow[kb + i];
                    b[i + 8] = (_Float16)wrow[kb + 16 + i];
                }
                Bf[g][c] = b;
            }
        }
    };

    auto loadA = [&](int mtbase, int c) -> v16h {
        int m  = mtbase + lidx;
        int kb = c * 32 + koff;
        v8h lo = *(const v8h*)&sH16[m][kb];
        v8h hi = *(const v8h*)&sH16[m][kb + 16];
        v16h a;
#pragma unroll
        for (int i = 0; i < 8; ++i) { a[i] = lo[i]; a[i + 8] = hi[i]; }
        return a;
    };

    float hn[2][8]; // new h values this lane produces (2 M-tiles x 8 rows)

    auto gru_step = [&]() {
        v8f aR0 = {0,0,0,0,0,0,0,0}, aZ0 = {0,0,0,0,0,0,0,0}, aN0 = {0,0,0,0,0,0,0,0};
        v8f aR1 = {0,0,0,0,0,0,0,0}, aZ1 = {0,0,0,0,0,0,0,0}, aN1 = {0,0,0,0,0,0,0,0};
#pragma unroll
        for (int c = 0; c < 4; ++c) {
            v16h a0 = loadA(0, c);
            v16h a1 = loadA(16, c);
            aR0 = __builtin_amdgcn_wmma_f32_16x16x32_f16(false, a0, false, Bf[0][c], (short)0, aR0, false, false);
            aZ0 = __builtin_amdgcn_wmma_f32_16x16x32_f16(false, a0, false, Bf[1][c], (short)0, aZ0, false, false);
            aN0 = __builtin_amdgcn_wmma_f32_16x16x32_f16(false, a0, false, Bf[2][c], (short)0, aN0, false, false);
            aR1 = __builtin_amdgcn_wmma_f32_16x16x32_f16(false, a1, false, Bf[0][c], (short)0, aR1, false, false);
            aZ1 = __builtin_amdgcn_wmma_f32_16x16x32_f16(false, a1, false, Bf[1][c], (short)0, aZ1, false, false);
            aN1 = __builtin_amdgcn_wmma_f32_16x16x32_f16(false, a1, false, Bf[2][c], (short)0, aN1, false, false);
        }
#pragma unroll
        for (int mt = 0; mt < 2; ++mt) {
#pragma unroll
            for (int k = 0; k < 8; ++k) {
                int row = mt * 16 + mh + k;
                float gir = biR, giz = biZ, gin = biN;
#pragma unroll
                for (int d = 0; d < DDIM; ++d) {
                    float xv = sX[row][d];
                    gir += xv * wR[d]; giz += xv * wZ[d]; gin += xv * wN[d];
                }
                float ghr = (mt ? aR1[k] : aR0[k]) + bhR;
                float ghz = (mt ? aZ1[k] : aZ0[k]) + bhZ;
                float ghn = (mt ? aN1[k] : aN0[k]) + bhN;
                float r = sigmoid_f(gir + ghr);
                float z = sigmoid_f(giz + ghz);
                float n = tanh_f(gin + r * ghn);
                float hold = sH32[row][jcol];
                hn[mt][k] = (1.0f - z) * n + z * hold;
            }
        }
    };

    auto writeH = [&]() {
#pragma unroll
        for (int mt = 0; mt < 2; ++mt)
#pragma unroll
            for (int k = 0; k < 8; ++k) {
                int row = mt * 16 + mh + k;
                sH32[row][jcol] = hn[mt][k];
                sH16[row][jcol] = (_Float16)hn[mt][k];
            }
    };

    // ---------------- encoder: 128 steps ----------------
    lane_setup(eWhh);
    for (int t = 0; t < T_OBS_N; ++t) {
        gru_step();
        __syncthreads();       // all reads of old h / x done
        writeH();
        if (t + 1 < T_OBS_N) {
            for (int i = tid; i < BTILE * DDIM; i += NTHREADS) {
                int r = i / DDIM, d = i - r * DDIM;
                sX[r][d] = obs[((size_t)(bbase + r) * T_OBS_N + (t + 1)) * DDIM + d];
            }
        }
        __syncthreads();       // new h / x visible
    }

    // ---------------- switch to decoder weights, x(0) = obs[:, -1, :] ------
    for (int i = tid; i < G3 * DDIM; i += NTHREADS) sWih[i] = cWih[i];
    for (int i = tid; i < G3; i += NTHREADS) { sBih[i] = cbih[i]; sBhh[i] = cbhh[i]; }
    for (int i = tid; i < BTILE * DDIM; i += NTHREADS) {
        int r = i / DDIM, d = i - r * DDIM;
        sX[r][d] = obs[((size_t)(bbase + r) * T_OBS_N + (T_OBS_N - 1)) * DDIM + d];
    }
    __syncthreads();
    lane_setup(cWhh);

    // ---------------- decoder: 64 steps, teacher forced + head -------------
    for (int t = 0; t < T_FUT_N; ++t) {
        gru_step();
        __syncthreads();
        writeH();
        if (t + 1 < T_FUT_N) {
            for (int i = tid; i < BTILE * DDIM; i += NTHREADS) {
                int r = i / DDIM, d = i - r * DDIM;
                sX[r][d] = target[((size_t)(bbase + r) * T_FUT_N + t) * DDIM + d];
            }
        }
        __syncthreads();       // new h visible -> head projection on h(t+1)
        for (int i = tid; i < BTILE * DDIM; i += NTHREADS) {
            int r = i / DDIM, d = i - r * DDIM;
            float acc = sHB[d];
#pragma unroll 8
            for (int j = 0; j < HDIM; ++j) acc += sH32[r][j] * sHW[d * HDIM + j];
            out[((size_t)(bbase + r) * T_FUT_N + t) * DDIM + d] = acc;
        }
    }
}

extern "C" void kernel_launch(void* const* d_in, const int* in_sizes, int n_in,
                              void* d_out, int out_size, void* d_ws, size_t ws_size,
                              hipStream_t stream) {
    const float* obs    = (const float*)d_in[0];
    const float* target = (const float*)d_in[1];
    const float* eWih   = (const float*)d_in[2];
    const float* eWhh   = (const float*)d_in[3];
    const float* ebih   = (const float*)d_in[4];
    const float* ebhh   = (const float*)d_in[5];
    const float* cWih   = (const float*)d_in[6];
    const float* cWhh   = (const float*)d_in[7];
    const float* cbih   = (const float*)d_in[8];
    const float* cbhh   = (const float*)d_in[9];
    const float* headW  = (const float*)d_in[10];
    const float* headb  = (const float*)d_in[11];
    float* out = (float*)d_out;

    dim3 grid(B_TOT / BTILE);   // 256 workgroups
    dim3 block(NTHREADS);       // 8 wave32 waves
    gru_seq2seq_kernel<<<grid, block, 0, stream>>>(
        obs, target, eWih, eWhh, ebih, ebhh,
        cWih, cWhh, cbih, cbhh, headW, headb, out);
}
